// GPT_90606630077042
// MI455X (gfx1250) — matrix operand order repaired and checked by reading, not compiled, not run on hardware
//
#include <hip/hip_runtime.h>
#include <math.h>

// ---------------- problem constants ----------------
#define B_   4
#define T_   1024
#define C_   768
#define NH_  12
#define HD_  64
#define L_   4
#define V_   512
#define M_   (B_ * T_)     // 4096 rows
#define FF_  (4 * C_)      // 3072

#define AS1 __attribute__((address_space(1)))
#define AS3 __attribute__((address_space(3)))

// ---------------- vector types ----------------
typedef __attribute__((ext_vector_type(16))) __bf16   bf16x16;
typedef __attribute__((ext_vector_type(8)))  float    f32x8;
typedef __attribute__((ext_vector_type(8)))  unsigned short u16x8;
typedef __attribute__((ext_vector_type(4)))  int      i32x4;

__device__ __forceinline__ unsigned short f2bf(float f) {
    unsigned int u = __float_as_uint(f);
    unsigned int r = u + 0x7FFFu + ((u >> 16) & 1u);   // round-to-nearest-even
    return (unsigned short)(r >> 16);
}

__device__ __forceinline__ f32x8 zero8() {
    f32x8 v;
#pragma unroll
    for (int i = 0; i < 8; ++i) v[i] = 0.0f;
    return v;
}

__device__ __forceinline__ f32x8 wmma_bf16(bf16x16 a, bf16x16 b, f32x8 c) {
    return __builtin_amdgcn_wmma_f32_16x16x32_bf16(
        /*neg_a=*/false, a, /*neg_b=*/false, b,
        /*c_mod=*/(short)0, c, /*reuse_a=*/false, /*reuse_b=*/false);
}

union FragU { bf16x16 v; u16x8 h[2]; unsigned short u[16]; };

// ---- CDNA5 async memory->LDS copy (16 bytes per lane) ----
__device__ __forceinline__ void async_ld16(void* lds, const void* gptr) {
#if defined(__has_builtin) && __has_builtin(__builtin_amdgcn_global_load_async_to_lds_b128)
    __builtin_amdgcn_global_load_async_to_lds_b128(
        (AS1 i32x4*)gptr, (AS3 i32x4*)lds, 0, 0);
#else
    unsigned int loff = (unsigned int)(size_t)(AS3 void*)lds;
    asm volatile("global_load_async_to_lds_b128 %0, %1, off"
                 :: "v"(loff), "v"(gptr) : "memory");
#endif
}

__device__ __forceinline__ void wait_async0() {
#if defined(__has_builtin) && __has_builtin(__builtin_amdgcn_s_wait_asynccnt)
    __builtin_amdgcn_s_wait_asynccnt(0);
#else
    asm volatile("s_wait_asynccnt 0x0" ::: "memory");
#endif
}

__device__ __forceinline__ void wait_async4() {
#if defined(__has_builtin) && __has_builtin(__builtin_amdgcn_s_wait_asynccnt)
    __builtin_amdgcn_s_wait_asynccnt(4);
#else
    asm volatile("s_wait_asynccnt 0x4" ::: "memory");
#endif
}

// A fragment (16x32 bf16, CDNA5 layout) from a row-major bf16 buffer, row stride ld.
// lane&15 = row; lane>>4 selects K-halves: runs [8h..8h+7] and [16+8h..16+8h+7].
__device__ __forceinline__ bf16x16 load_a_frag(const unsigned short* base, int ld, int lane) {
    int r = lane & 15, hf = lane >> 4;
    const unsigned short* p = base + (size_t)r * ld;
    FragU f;
    f.h[0] = *(const u16x8*)(p + 8 * hf);
    f.h[1] = *(const u16x8*)(p + 16 + 8 * hf);
    return f.v;
}

// ---------------- simple elementwise kernels ----------------
__global__ void cvt_f32_bf16(const float* __restrict__ s, unsigned short* __restrict__ d, int n) {
    int i = blockIdx.x * blockDim.x + threadIdx.x;
    if (i < n) d[i] = f2bf(s[i]);
}

__global__ void embed_kernel(const int* __restrict__ idx, const float* __restrict__ tok,
                             const float* __restrict__ pos, const float* __restrict__ type1,
                             float* __restrict__ x) {
    int i = blockIdx.x * blockDim.x + threadIdx.x;   // over M_*C_
    int c = i % C_, m = i / C_;
    int t = m % T_;
    x[i] = tok[(size_t)idx[m] * C_ + c] + pos[(size_t)t * C_ + c] + type1[c];
}

// LayerNorm: one block per row; writes bf16 row for GEMM-A consumption.
__global__ void __launch_bounds__(256) ln_kernel(const float* __restrict__ x,
                                                 const float* __restrict__ s,
                                                 const float* __restrict__ b,
                                                 unsigned short* __restrict__ o) {
    __shared__ float red[256];
    int m = blockIdx.x, tid = threadIdx.x;
    const float* xr = x + (size_t)m * C_;
    float sum = 0.f, sq = 0.f;
    for (int c = tid; c < C_; c += 256) { float v = xr[c]; sum += v; sq += v * v; }
    red[tid] = sum; __syncthreads();
    for (int off = 128; off > 0; off >>= 1) { if (tid < off) red[tid] += red[tid + off]; __syncthreads(); }
    float mean = red[0] / (float)C_;
    __syncthreads();
    red[tid] = sq; __syncthreads();
    for (int off = 128; off > 0; off >>= 1) { if (tid < off) red[tid] += red[tid + off]; __syncthreads(); }
    float var = red[0] / (float)C_ - mean * mean;
    float rstd = rsqrtf(var + 1e-5f);
    unsigned short* orow = o + (size_t)m * C_;
    for (int c = tid; c < C_; c += 256)
        orow[c] = f2bf((xr[c] - mean) * rstd * s[c] + b[c]);
}

// ---------------- generic bf16 WMMA GEMM ----------------
// D[M,N] = A[M,K] (bf16, lda) @ W[K,N] (bf16, ldb) + bias
// Double-buffered async (memory->LDS DMA) pipeline: issue tile i+1 while computing tile i.
// OMODE: 0 = f32 store, 1 = f32 residual add, 2 = bf16 store, 3 = bf16 gelu store
template <int OMODE>
__global__ void __launch_bounds__(128) gemm_bf16(
    const unsigned short* __restrict__ A, int lda,
    const unsigned short* __restrict__ W, int ldb,
    const float* __restrict__ bias,
    float* __restrict__ outF, unsigned short* __restrict__ outH, int ldo,
    int N, int K)
{
    __shared__ unsigned short As[2][64 * 32];
    __shared__ unsigned short Bs[2][32 * 64];
    int tid = threadIdx.x;
    int lane = tid & 31, w = tid >> 5;
    int wm = (w >> 1) * 32, wn = (w & 1) * 32;
    int row0 = blockIdx.y * 64, col0 = blockIdx.x * 64;

    // staging coordinates (16 elements = 32B per thread per tile, as 2 async 16B ops)
    int ra = tid >> 1, ca = (tid & 1) * 16;
    int rb = tid >> 2, cb = (tid & 3) * 16;

    f32x8 acc[2][2];
#pragma unroll
    for (int i = 0; i < 2; ++i)
#pragma unroll
        for (int j = 0; j < 2; ++j) acc[i][j] = zero8();

    auto issue = [&](int kk, int buf) {
        const unsigned short* ga = A + (size_t)(row0 + ra) * lda + kk + ca;
        const unsigned short* gb = W + (size_t)(kk + rb) * ldb + col0 + cb;
        async_ld16(&As[buf][ra * 32 + ca],     ga);
        async_ld16(&As[buf][ra * 32 + ca + 8], ga + 8);
        async_ld16(&Bs[buf][rb * 64 + cb],     gb);
        async_ld16(&Bs[buf][rb * 64 + cb + 8], gb + 8);
    };

    int nk = K >> 5;            // K/32 steps
    issue(0, 0);
    for (int i = 0; i < nk; ++i) {
        if (i + 1 < nk) {       // uniform scalar branch
            issue((i + 1) << 5, (i + 1) & 1);
            wait_async4();      // older 4 ops (current buffer) retired, newer 4 in flight
        } else {
            wait_async0();
        }
        __syncthreads();

        int buf = i & 1;
        bf16x16 a0 = load_a_frag(&As[buf][(wm) * 32], 32, lane);
        bf16x16 a1 = load_a_frag(&As[buf][(wm + 16) * 32], 32, lane);
        int n16 = lane & 15, hf = lane >> 4;
        FragU b0, b1;
#pragma unroll
        for (int j = 0; j < 16; ++j) {
            int kr = (j < 8 ? 8 * hf + j : 16 + 8 * hf + (j - 8)) * 64;
            b0.u[j] = Bs[buf][kr + wn + n16];
            b1.u[j] = Bs[buf][kr + wn + 16 + n16];
        }
        acc[0][0] = wmma_bf16(a0, b0.v, acc[0][0]);
        acc[0][1] = wmma_bf16(a0, b1.v, acc[0][1]);
        acc[1][0] = wmma_bf16(a1, b0.v, acc[1][0]);
        acc[1][1] = wmma_bf16(a1, b1.v, acc[1][1]);
        __syncthreads();        // protect buffer reuse two iterations ahead
    }

    int n16 = lane & 15, hf = lane >> 4;
#pragma unroll
    for (int ti = 0; ti < 2; ++ti)
#pragma unroll
        for (int tj = 0; tj < 2; ++tj) {
#pragma unroll
            for (int g = 0; g < 8; ++g) {
                int r = row0 + wm + ti * 16 + g + 8 * hf;
                int c = col0 + wn + tj * 16 + n16;
                float val = acc[ti][tj][g];
                if (bias) val += bias[c];
                size_t o = (size_t)r * ldo + c;
                if (OMODE == 0)      outF[o] = val;
                else if (OMODE == 1) outF[o] += val;
                else if (OMODE == 2) outH[o] = f2bf(val);
                else {
                    float gl = 0.5f * val * (1.0f + erff(val * 0.70710678118654752f));
                    outH[o] = f2bf(gl);
                }
            }
        }
}

// ---------------- attention: scores + softmax ----------------
// one block per (b, h, 16-row q-tile); 8 waves each cover 128 k-columns
__global__ void __launch_bounds__(256) attn_scores(
    const unsigned short* __restrict__ qb, const unsigned short* __restrict__ kb,
    float* __restrict__ att /* layer base: [B][NH][T][T] */)
{
    extern __shared__ float sc[];   // 16 * T_ floats (64 KB)
    int blk = blockIdx.x;
    int qt = blk & (T_ / 16 - 1);
    int h  = (blk >> 6) % NH_;
    int b  = blk / ((T_ / 16) * NH_);
    int tid = threadIdx.x, lane = tid & 31;
    // wave index as an SGPR so causal-skip branches are scalar (EXEC stays all-ones for WMMA)
    int w = __builtin_amdgcn_readfirstlane(tid >> 5);
    int r16 = lane & 15, hf = lane >> 4;

    const unsigned short* qbase = qb + ((size_t)(b * T_ + qt * 16) * C_) + h * HD_;
    const unsigned short* kbase = kb + ((size_t)b * T_) * C_ + h * HD_;
    int qmax = qt * 16 + 15;

    f32x8 acc[8];
#pragma unroll
    for (int i = 0; i < 8; ++i) acc[i] = zero8();

    for (int kk = 0; kk < HD_; kk += 32) {
        bf16x16 afr = load_a_frag(qbase + kk, C_, lane);
#pragma unroll
        for (int nt = 0; nt < 8; ++nt) {
            int cbase = w * 128 + nt * 16;          // scalar
            if (cbase > qmax) continue;             // scalar branch; fully-masked tile
            FragU fb;
            const unsigned short* kp = kbase + (size_t)(cbase + r16) * C_ + kk + 8 * hf;
            fb.h[0] = *(const u16x8*)(kp);
            fb.h[1] = *(const u16x8*)(kp + 16);
            acc[nt] = wmma_bf16(afr, fb.v, acc[nt]);
        }
    }

    const float scale = 0.125f;   // 1/sqrt(64)
#pragma unroll
    for (int nt = 0; nt < 8; ++nt) {
        int cbase = w * 128 + nt * 16;
#pragma unroll
        for (int g = 0; g < 8; ++g) {
            int rq = g + 8 * hf;
            int qi = qt * 16 + rq;
            int kj = cbase + r16;
            float v = acc[nt][g] * scale;
            if (kj > qi) v = -__builtin_inff();
            sc[rq * T_ + kj] = v;
        }
    }
    __syncthreads();

    // softmax: row = tid/16 (0..15), strided over T_ by 16 lanes
    int row = tid >> 4, li = tid & 15;
    float mx = -__builtin_inff();
    for (int c = li; c < T_; c += 16) mx = fmaxf(mx, sc[row * T_ + c]);
#pragma unroll
    for (int o = 8; o > 0; o >>= 1) mx = fmaxf(mx, __shfl_xor(mx, o, 16));
    float sm = 0.f;
    for (int c = li; c < T_; c += 16) sm += expf(sc[row * T_ + c] - mx);
#pragma unroll
    for (int o = 8; o > 0; o >>= 1) sm += __shfl_xor(sm, o, 16);
    float inv = 1.0f / sm;
    size_t obase = (((size_t)b * NH_ + h) * T_ + (qt * 16 + row)) * T_;
    for (int c = li; c < T_; c += 16)
        att[obase + c] = expf(sc[row * T_ + c] - mx) * inv;
}

// ---------------- attention: y = att @ V ----------------
// one block = 8 waves; each wave computes a 16x64 (one q-tile x full head) output
__global__ void __launch_bounds__(256) attn_av(
    const float* __restrict__ att /* layer base */, const unsigned short* __restrict__ vb,
    unsigned short* __restrict__ y /* [M,C] bf16 */)
{
    int blk = blockIdx.x;           // B_*NH_*8
    int sb = blk & 7;
    int h  = (blk >> 3) % NH_;
    int b  = blk / (8 * NH_);
    int tid = threadIdx.x, lane = tid & 31;
    int w = __builtin_amdgcn_readfirstlane(tid >> 5);   // scalar wave index
    int qt = sb * 8 + w;            // scalar: 0..63
    int r16 = lane & 15, hf = lane >> 4;

    const float* abase = att + (((size_t)b * NH_ + h) * T_ + qt * 16) * T_;
    const unsigned short* vbase = vb + ((size_t)b * T_) * C_ + h * HD_;

    f32x8 acc[4];
#pragma unroll
    for (int i = 0; i < 4; ++i) acc[i] = zero8();

    int kmax = ((qt * 16 + 16) + 31) & ~31;   // scalar causal bound
    if (kmax > T_) kmax = T_;

    for (int kk = 0; kk < kmax; kk += 32) {   // scalar loop -> EXEC stays all-ones
        if (kk + 32 < kmax) {                 // prefetch next K-step (global_prefetch_b8)
            __builtin_prefetch(abase + (size_t)r16 * T_ + kk + 32, 0, 1);
            __builtin_prefetch(vbase + (size_t)(kk + 32 + 16 * hf) * C_ + r16, 0, 1);
        }
        FragU fa;
        const float* ap = abase + (size_t)r16 * T_ + kk;
#pragma unroll
        for (int j = 0; j < 8; ++j) fa.u[j] = f2bf(ap[8 * hf + j]);
#pragma unroll
        for (int j = 0; j < 8; ++j) fa.u[8 + j] = f2bf(ap[16 + 8 * hf + j]);
#pragma unroll
        for (int nt = 0; nt < 4; ++nt) {
            FragU fb;
#pragma unroll
            for (int j = 0; j < 16; ++j)
                fb.u[j] = vbase[(size_t)(kk + (j < 8 ? 8 * hf + j : 16 + 8 * hf + (j - 8))) * C_ + nt * 16 + r16];
            acc[nt] = wmma_bf16(fa.v, fb.v, acc[nt]);
        }
    }

#pragma unroll
    for (int nt = 0; nt < 4; ++nt)
#pragma unroll
        for (int g = 0; g < 8; ++g) {
            int m = b * T_ + qt * 16 + g + 8 * hf;
            y[(size_t)m * C_ + h * HD_ + nt * 16 + r16] = f2bf(acc[nt][g]);
        }
}

// ---------------- host side ----------------
static void launch_gemm(int mode, const unsigned short* A, int lda,
                        const unsigned short* W, int ldb, const float* bias,
                        float* oF, unsigned short* oH, int ldo, int N, int K,
                        hipStream_t s) {
    dim3 grid(N / 64, M_ / 64);
    switch (mode) {
    case 0: gemm_bf16<0><<<grid, 128, 0, s>>>(A, lda, W, ldb, bias, oF, oH, ldo, N, K); break;
    case 1: gemm_bf16<1><<<grid, 128, 0, s>>>(A, lda, W, ldb, bias, oF, oH, ldo, N, K); break;
    case 2: gemm_bf16<2><<<grid, 128, 0, s>>>(A, lda, W, ldb, bias, oF, oH, ldo, N, K); break;
    default: gemm_bf16<3><<<grid, 128, 0, s>>>(A, lda, W, ldb, bias, oF, oH, ldo, N, K); break;
    }
}

extern "C" void kernel_launch(void* const* d_in, const int* in_sizes, int n_in,
                              void* d_out, int out_size, void* d_ws, size_t ws_size,
                              hipStream_t stream) {
    (void)in_sizes; (void)n_in; (void)out_size; (void)ws_size;

    const int*   idx    = (const int*)d_in[0];
    const float* tok    = (const float*)d_in[1];
    const float* typeE  = (const float*)d_in[2];
    const float* pos    = (const float*)d_in[3];
    const float* ln1_s  = (const float*)d_in[4];
    const float* ln1_b  = (const float*)d_in[5];
    const float* ln2_s  = (const float*)d_in[6];
    const float* ln2_b  = (const float*)d_in[7];
    const float* wq     = (const float*)d_in[8];
    const float* bq     = (const float*)d_in[9];
    const float* wk     = (const float*)d_in[10];
    const float* bk     = (const float*)d_in[11];
    const float* wv     = (const float*)d_in[12];
    const float* bv     = (const float*)d_in[13];
    const float* wp     = (const float*)d_in[14];
    const float* bp     = (const float*)d_in[15];
    const float* w1     = (const float*)d_in[16];
    const float* b1     = (const float*)d_in[17];
    const float* w2     = (const float*)d_in[18];
    const float* b2     = (const float*)d_in[19];
    const float* lnf_s  = (const float*)d_in[20];
    const float* lnf_b  = (const float*)d_in[21];
    const float* head_w = (const float*)d_in[22];

    // workspace carve (256B aligned)
    char* p = (char*)d_ws;
    auto alloc = [&](size_t bytes) -> void* {
        void* r = (void*)p;
        p += (bytes + 255) & ~(size_t)255;
        return r;
    };
    float* x            = (float*)alloc((size_t)M_ * C_ * 4);
    unsigned short* abuf = (unsigned short*)alloc((size_t)M_ * C_ * 2);
    unsigned short* mbuf = (unsigned short*)alloc((size_t)M_ * FF_ * 2);
    unsigned short* qb   = (unsigned short*)alloc((size_t)M_ * C_ * 2);
    unsigned short* kbuf = (unsigned short*)alloc((size_t)M_ * C_ * 2);
    unsigned short* vbuf = (unsigned short*)alloc((size_t)M_ * C_ * 2);
    unsigned short* wqB  = (unsigned short*)alloc((size_t)L_ * C_ * C_ * 2);
    unsigned short* wkB  = (unsigned short*)alloc((size_t)L_ * C_ * C_ * 2);
    unsigned short* wvB  = (unsigned short*)alloc((size_t)L_ * C_ * C_ * 2);
    unsigned short* wpB  = (unsigned short*)alloc((size_t)L_ * C_ * C_ * 2);
    unsigned short* w1B  = (unsigned short*)alloc((size_t)L_ * C_ * FF_ * 2);
    unsigned short* w2B  = (unsigned short*)alloc((size_t)L_ * FF_ * C_ * 2);
    unsigned short* hB   = (unsigned short*)alloc((size_t)C_ * V_ * 2);

    const int TPB = 256;
    auto cdiv = [](long long a, long long b) { return (int)((a + b - 1) / b); };

    // weight conversion to bf16
    cvt_f32_bf16<<<cdiv((long long)L_ * C_ * C_, TPB), TPB, 0, stream>>>(wq, wqB, L_ * C_ * C_);
    cvt_f32_bf16<<<cdiv((long long)L_ * C_ * C_, TPB), TPB, 0, stream>>>(wk, wkB, L_ * C_ * C_);
    cvt_f32_bf16<<<cdiv((long long)L_ * C_ * C_, TPB), TPB, 0, stream>>>(wv, wvB, L_ * C_ * C_);
    cvt_f32_bf16<<<cdiv((long long)L_ * C_ * C_, TPB), TPB, 0, stream>>>(wp, wpB, L_ * C_ * C_);
    cvt_f32_bf16<<<cdiv((long long)L_ * C_ * FF_, TPB), TPB, 0, stream>>>(w1, w1B, L_ * C_ * FF_);
    cvt_f32_bf16<<<cdiv((long long)L_ * FF_ * C_, TPB), TPB, 0, stream>>>(w2, w2B, L_ * FF_ * C_);
    cvt_f32_bf16<<<cdiv((long long)C_ * V_, TPB), TPB, 0, stream>>>(head_w, hB, C_ * V_);

    // embeddings: tok + pos + type[1]
    embed_kernel<<<cdiv((long long)M_ * C_, TPB), TPB, 0, stream>>>(idx, tok, pos, typeE + C_, x);

    float* logits   = (float*)d_out;
    float* attn_all = (float*)d_out + (size_t)M_ * V_;

    for (int l = 0; l < L_; ++l) {
        size_t wOff  = (size_t)l * C_ * C_;
        size_t w1Off = (size_t)l * C_ * FF_;

        ln_kernel<<<M_, 256, 0, stream>>>(x, ln1_s + l * C_, ln1_b + l * C_, abuf);
        launch_gemm(2, abuf, C_, wqB + wOff, C_, bq + l * C_, nullptr, qb,   C_, C_, C_, stream);
        launch_gemm(2, abuf, C_, wkB + wOff, C_, bk + l * C_, nullptr, kbuf, C_, C_, C_, stream);
        launch_gemm(2, abuf, C_, wvB + wOff, C_, bv + l * C_, nullptr, vbuf, C_, C_, C_, stream);

        float* attl = attn_all + (size_t)l * B_ * NH_ * T_ * T_;
        attn_scores<<<B_ * NH_ * (T_ / 16), 256, 16 * T_ * sizeof(float), stream>>>(qb, kbuf, attl);
        attn_av<<<B_ * NH_ * 8, 256, 0, stream>>>(attl, vbuf, abuf);

        launch_gemm(1, abuf, C_, wpB + wOff, C_, bp + l * C_, x, nullptr, C_, C_, C_, stream);

        ln_kernel<<<M_, 256, 0, stream>>>(x, ln2_s + l * C_, ln2_b + l * C_, abuf);
        launch_gemm(3, abuf, C_, w1B + w1Off, FF_, b1 + l * FF_, nullptr, mbuf, FF_, FF_, C_, stream);
        launch_gemm(1, mbuf, FF_, w2B + w1Off, C_, b2 + l * C_, x, nullptr, C_, C_, FF_, stream);
    }

    ln_kernel<<<M_, 256, 0, stream>>>(x, lnf_s, lnf_b, abuf);
    launch_gemm(0, abuf, C_, hB, V_, nullptr, logits, nullptr, V_, V_, C_, stream);
}
